// BidirectionalNeuralMemory_3925600109253
// MI455X (gfx1250) — matrix operand order repaired and checked by reading, not compiled, not run on hardware
//
#include <hip/hip_runtime.h>
#include <math.h>

// ---------------------------------------------------------------------------
// Types
// ---------------------------------------------------------------------------
typedef __bf16  v16bf __attribute__((ext_vector_type(16)));
typedef float   v8f   __attribute__((ext_vector_type(8)));
typedef unsigned int u32x4 __attribute__((ext_vector_type(4)));

union V16U { v16bf v; struct { u32x4 lo, hi; } q; };

__device__ __forceinline__ __bf16 f2bf(float f) {
  unsigned u = __builtin_bit_cast(unsigned, f);
  unsigned r = u + 0x7FFFu + ((u >> 16) & 1u);           // round-to-nearest-even
  unsigned short h = (unsigned short)(r >> 16);
  return __builtin_bit_cast(__bf16, h);
}
__device__ __forceinline__ float bf2f(__bf16 b) {
  unsigned short h = __builtin_bit_cast(unsigned short, b);
  unsigned u = ((unsigned)h) << 16;
  return __builtin_bit_cast(float, u);
}

// ---------------------------------------------------------------------------
// Problem constants
// ---------------------------------------------------------------------------
constexpr int BSZ  = 8;
constexpr int NSEQ = 2048;
constexpr int DIM  = 768;
constexpr int NH   = 8;
constexpr int DH   = 96;
constexpr int HID  = 384;
constexpr int ROWS = BSZ * NSEQ;      // 16384
constexpr float CLAMPV = 100.0f;

// ---------------------------------------------------------------------------
// Weight conversion kernels (f32 -> bf16, optionally transposed)
// ---------------------------------------------------------------------------
__global__ void tr_cvt_kernel(const float* __restrict__ W, __bf16* __restrict__ out,
                              int R, int C) {
  int idx = blockIdx.x * 256 + threadIdx.x;
  if (idx < R * C) {
    int r = idx / C, c = idx % C;
    out[(size_t)c * R + r] = f2bf(W[idx]);     // out[c][r] = W[r][c]
  }
}
__global__ void cvt_kernel(const float* __restrict__ W, __bf16* __restrict__ out, int n) {
  int idx = blockIdx.x * 256 + threadIdx.x;
  if (idx < n) out[idx] = f2bf(W[idx]);
}

// ---------------------------------------------------------------------------
// Fused LayerNorm (store + retrieve affine) + adaptive-lr step head
// one block (256 threads = 8 waves) per row of 768
// ---------------------------------------------------------------------------
__global__ __launch_bounds__(256)
void ln_kernel(const float* __restrict__ seq,
               const float* __restrict__ gs, const float* __restrict__ bs,
               const float* __restrict__ gr, const float* __restrict__ br,
               const float* __restrict__ Wstep, const float* __restrict__ bstep,
               __bf16* __restrict__ s_bf, __bf16* __restrict__ r_bf,
               float* __restrict__ lr) {
  __shared__ float red[256];
  __shared__ float srow[DIM];
  const int row = blockIdx.x;
  const int t = threadIdx.x;
  const float* x = seq + (size_t)row * DIM;

  float v0 = x[t], v1 = x[t + 256], v2 = x[t + 512];
  red[t] = v0 + v1 + v2;
  __syncthreads();
  for (int s = 128; s > 0; s >>= 1) { if (t < s) red[t] += red[t + s]; __syncthreads(); }
  const float mean = red[0] * (1.0f / DIM);
  __syncthreads();
  float d0 = v0 - mean, d1 = v1 - mean, d2 = v2 - mean;
  red[t] = d0 * d0 + d1 * d1 + d2 * d2;
  __syncthreads();
  for (int s = 128; s > 0; s >>= 1) { if (t < s) red[t] += red[t + s]; __syncthreads(); }
  const float rstd = rsqrtf(red[0] * (1.0f / DIM) + 1e-5f);

  float n0 = d0 * rstd, n1 = d1 * rstd, n2 = d2 * rstd;
  float s0 = n0 * gs[t]       + bs[t];
  float s1 = n1 * gs[t + 256] + bs[t + 256];
  float s2 = n2 * gs[t + 512] + bs[t + 512];
  srow[t] = s0; srow[t + 256] = s1; srow[t + 512] = s2;
  __bf16* so = s_bf + (size_t)row * DIM;
  so[t] = f2bf(s0); so[t + 256] = f2bf(s1); so[t + 512] = f2bf(s2);
  __bf16* ro = r_bf + (size_t)row * DIM;
  ro[t]       = f2bf(n0 * gr[t]       + br[t]);
  ro[t + 256] = f2bf(n1 * gr[t + 256] + br[t + 256]);
  ro[t + 512] = f2bf(n2 * gr[t + 512] + br[t + 512]);
  __syncthreads();

  // adaptive lr: head h handled by wave h (wave32, 8 waves)
  const int wv = t >> 5, ln = t & 31;
  float p = 0.0f;
  for (int i = ln; i < DIM; i += 32) p += srow[i] * Wstep[i * NH + wv];
  for (int off = 16; off > 0; off >>= 1) p += __shfl_xor(p, off, 32);
  if (ln == 0)
    lr[(size_t)row * NH + wv] = 0.1f / (1.0f + __expf(-(p + bstep[wv])));
}

// ---------------------------------------------------------------------------
// Generic strided bf16 WMMA GEMM with fused epilogues.
//   C[m,n] = sum_k A(m,k) * B(k,n), per batch z (z>>3, z&7 stride split
//   lets [b, s, h*96+d] head-split tensors act as batched operands).
// One wave computes a 32x32 tile (2x2 grid of 16x16 WMMA accumulators);
// 4 waves/block tile along N. Register-level reuse: each A/B fragment
// feeds two WMMAs.
// ---------------------------------------------------------------------------
enum { EPI_BF = 0, EPI_F32 = 1, EPI_CLIP = 2, EPI_WE = 3, EPI_WSUBT = 4, EPI_GELU = 5 };

template <int EPI>
__device__ __forceinline__ void epi_tile(v8f acc, int mbase, int ncol, int zb, int zh,
                                         void* Cp, size_t cB, size_t cH, int ldc,
                                         const __bf16* x0, size_t x0B, size_t x0H, int ldx0,
                                         const float* x1, size_t x1B, size_t x1H) {
  if constexpr (EPI == EPI_WSUBT) {
    __bf16* C = (__bf16*)Cp + (size_t)zb * cB + (size_t)zh * cH;
#pragma unroll
    for (int r = 0; r < 8; ++r) {
      size_t idx = (size_t)ncol * (size_t)ldc + (size_t)(mbase + r);
      C[idx] = f2bf(bf2f(x0[idx]) - acc[r]);   // w_new = w - gw, stored transposed
    }
  } else if constexpr (EPI == EPI_F32) {
    float* C = (float*)Cp + (size_t)zb * cB + (size_t)zh * cH;
#pragma unroll
    for (int r = 0; r < 8; ++r)
      C[(size_t)(mbase + r) * (size_t)ldc + ncol] = acc[r];
  } else {
    __bf16* C = (__bf16*)Cp + (size_t)zb * cB + (size_t)zh * cH;
#pragma unroll
    for (int r = 0; r < 8; ++r) {
      float c = acc[r];
      if constexpr (EPI == EPI_CLIP) {
        c = fminf(fmaxf(c, -CLAMPV), CLAMPV);
      } else if constexpr (EPI == EPI_GELU) {
        c = 0.5f * c * (1.0f + erff(c * 0.70710678118654752f));
      } else if constexpr (EPI == EPI_WE) {
        c = fminf(fmaxf(c, -CLAMPV), CLAMPV);
        float v = bf2f(x0[(size_t)zb * x0B + (size_t)zh * x0H +
                          (size_t)(mbase + r) * (size_t)ldx0 + ncol]);
        float l = x1[(size_t)zb * x1B + (size_t)zh * x1H + (size_t)(mbase + r) * NH];
        c = (c - v) * l;
      }
      C[(size_t)(mbase + r) * (size_t)ldc + ncol] = f2bf(c);
    }
  }
}

template <bool AV, bool BV, int EPI>
__global__ __launch_bounds__(128)
void gemm_wmma(const __bf16* __restrict__ A, size_t aB, size_t aH, int sAm, int sAk,
               const __bf16* __restrict__ B, size_t bB, size_t bH, int sBk, int sBn,
               void* __restrict__ Cp, size_t cB, size_t cH, int ldc,
               int N, int K,
               const __bf16* __restrict__ x0, size_t x0B, size_t x0H, int ldx0,
               const float* __restrict__ x1, size_t x1B, size_t x1H) {
  const int z  = blockIdx.z;
  const int zb = z >> 3, zh = z & 7;
  A += (size_t)zb * aB + (size_t)zh * aH;
  B += (size_t)zb * bB + (size_t)zh * bH;

  const int wave = threadIdx.x >> 5;
  const int lane = threadIdx.x & 31;
  const int n0 = (blockIdx.y * 4 + wave) * 32;
  if (n0 >= N) return;                         // wave-uniform: EXEC stays full

  const int m0 = blockIdx.x * 32;
  const int half = lane >> 4, l16 = lane & 15;
  const int akoff = half * 16;                 // A: 16 consecutive K per lane
  const int bkoff = half * 16;                 // B: 16 consecutive K per lane

  const size_t arow0 = (size_t)(m0 + l16) * (size_t)sAm;
  const size_t arow1 = arow0 + (size_t)16 * (size_t)sAm;
  const size_t bcol0 = (size_t)(n0 + l16) * (size_t)sBn;
  const size_t bcol1 = bcol0 + (size_t)16 * (size_t)sBn;

  v8f acc00 = {0.f, 0.f, 0.f, 0.f, 0.f, 0.f, 0.f, 0.f};
  v8f acc01 = acc00, acc10 = acc00, acc11 = acc00;

  for (int k0 = 0; k0 < K; k0 += 32) {
    V16U a0, a1, b0, b1;
    if constexpr (AV) {                        // sAk == 1 fast path (32B aligned tiles)
      const __bf16* p0 = A + arow0 + (size_t)(k0 + akoff);
      const __bf16* p1 = A + arow1 + (size_t)(k0 + akoff);
      a0.q.lo = *(const u32x4*)p0;  a0.q.hi = *(const u32x4*)(p0 + 8);
      a1.q.lo = *(const u32x4*)p1;  a1.q.hi = *(const u32x4*)(p1 + 8);
      if (k0 + 64 < K) {                       // gfx1250 global_prefetch, near-cache policy
        __builtin_prefetch(p0 + 64, 0, 3);
        __builtin_prefetch(p1 + 64, 0, 3);
      }
    } else {
#pragma unroll
      for (int j = 0; j < 8; ++j) {
        size_t klo = (size_t)(k0 + akoff + j) * (size_t)sAk;
        size_t khi = (size_t)(k0 + akoff + 8 + j) * (size_t)sAk;
        a0.v[j] = A[arow0 + klo];  a0.v[j + 8] = A[arow0 + khi];
        a1.v[j] = A[arow1 + klo];  a1.v[j + 8] = A[arow1 + khi];
      }
    }
    if constexpr (BV) {                        // sBk == 1 fast path
      const __bf16* p0 = B + bcol0 + (size_t)(k0 + bkoff);
      const __bf16* p1 = B + bcol1 + (size_t)(k0 + bkoff);
      b0.q.lo = *(const u32x4*)p0;  b0.q.hi = *(const u32x4*)(p0 + 8);
      b1.q.lo = *(const u32x4*)p1;  b1.q.hi = *(const u32x4*)(p1 + 8);
      if (k0 + 64 < K) {
        __builtin_prefetch(p0 + 64, 0, 3);
        __builtin_prefetch(p1 + 64, 0, 3);
      }
    } else {
#pragma unroll
      for (int j = 0; j < 16; ++j) {
        size_t kk = (size_t)(k0 + bkoff + j) * (size_t)sBk;
        b0.v[j] = B[kk + bcol0];
        b1.v[j] = B[kk + bcol1];
      }
    }
    acc00 = __builtin_amdgcn_wmma_f32_16x16x32_bf16(false, a0.v, false, b0.v,
                                                    (short)0, acc00, false, false);
    acc01 = __builtin_amdgcn_wmma_f32_16x16x32_bf16(false, a0.v, false, b1.v,
                                                    (short)0, acc01, false, false);
    acc10 = __builtin_amdgcn_wmma_f32_16x16x32_bf16(false, a1.v, false, b0.v,
                                                    (short)0, acc10, false, false);
    acc11 = __builtin_amdgcn_wmma_f32_16x16x32_bf16(false, a1.v, false, b1.v,
                                                    (short)0, acc11, false, false);
  }

  // C layout per 16x16 tile: VGPR r -> M = r + 8*half, N = l16
  const int mb0 = m0 + 8 * half, mb1 = m0 + 16 + 8 * half;
  const int nc0 = n0 + l16, nc1 = n0 + 16 + l16;
  epi_tile<EPI>(acc00, mb0, nc0, zb, zh, Cp, cB, cH, ldc, x0, x0B, x0H, ldx0, x1, x1B, x1H);
  epi_tile<EPI>(acc01, mb0, nc1, zb, zh, Cp, cB, cH, ldc, x0, x0B, x0H, ldx0, x1, x1B, x1H);
  epi_tile<EPI>(acc10, mb1, nc0, zb, zh, Cp, cB, cH, ldc, x0, x0B, x0H, ldx0, x1, x1B, x1H);
  epi_tile<EPI>(acc11, mb1, nc1, zb, zh, Cp, cB, cH, ldc, x0, x0B, x0H, ldx0, x1, x1B, x1H);
}

// ---------------------------------------------------------------------------
// Host orchestration
// ---------------------------------------------------------------------------
extern "C" void kernel_launch(void* const* d_in, const int* in_sizes, int n_in,
                              void* d_out, int out_size, void* d_ws, size_t ws_size,
                              hipStream_t stream) {
  const float* seq     = (const float*)d_in[0];
  const float* g_store = (const float*)d_in[1];
  const float* b_store = (const float*)d_in[2];
  const float* g_ret   = (const float*)d_in[3];
  const float* b_ret   = (const float*)d_in[4];
  const float* Wq      = (const float*)d_in[5];
  const float* Wk      = (const float*)d_in[6];
  const float* Wv      = (const float*)d_in[7];
  const float* Wsplit  = (const float*)d_in[8];
  const float* Wmerge  = (const float*)d_in[9];
  const float* mem_w0  = (const float*)d_in[10];
  const float* mem_w1  = (const float*)d_in[11];
  const float* W_step  = (const float*)d_in[12];
  const float* b_step  = (const float*)d_in[13];
  float* out = (float*)d_out;

  // ---- workspace layout (256B aligned) ----
  char* wsc = (char*)d_ws;
  size_t off = 0;
  auto take = [&](size_t bytes) -> char* {
    char* p = wsc + off;
    off += (bytes + 255) & ~(size_t)255;
    return p;
  };
  const size_t RD = (size_t)ROWS * DIM;              // 16384*768
  __bf16* s_bf   = (__bf16*)take(RD * 2);
  __bf16* r_bf   = (__bf16*)take(RD * 2);
  __bf16* tmp_bf = (__bf16*)take(RD * 2);
  __bf16* kproj  = (__bf16*)take(RD * 2);
  __bf16* vproj  = (__bf16*)take(RD * 2);
  float*  lr     = (float*) take((size_t)ROWS * NH * 4);
  __bf16* WqT    = (__bf16*)take((size_t)DIM * DIM * 2);
  __bf16* WkT    = (__bf16*)take((size_t)DIM * DIM * 2);
  __bf16* WvT    = (__bf16*)take((size_t)DIM * DIM * 2);
  __bf16* WspT   = (__bf16*)take((size_t)DIM * DIM * 2);
  __bf16* WmgT   = (__bf16*)take((size_t)DIM * DIM * 2);
  __bf16* w0t    = (__bf16*)take((size_t)DH * HID * 2);   // [HID][DH]
  __bf16* w1t    = (__bf16*)take((size_t)DH * HID * 2);   // [DH][HID]
  __bf16* w1n    = (__bf16*)take((size_t)DH * HID * 2);   // [HID][DH] plain
  __bf16* w0newT = (__bf16*)take((size_t)64 * HID * DH * 2);
  __bf16* w1newT = (__bf16*)take((size_t)64 * DH * HID * 2);
  __bf16* pred0  = (__bf16*)take((size_t)64 * NSEQ * HID * 2);
  __bf16* we     = (__bf16*)take((size_t)64 * NSEQ * DH * 2);
  __bf16* dh0    = (__bf16*)take((size_t)64 * NSEQ * HID * 2);
  __bf16* qproj  = s_bf;    // reuse: s last read by v-chain
  __bf16* h_bf   = dh0;     // reuse: dh0 last read by gw0
  __bf16* merged = pred0;   // reuse: pred0 last read by gw1

  const dim3 blk(128);
  const size_t WG = 36864;  // per-batch w_new elems (96*384)

  // ---- weight conversions ----
  {
    int n = DIM * DIM, g = (n + 255) / 256;
    tr_cvt_kernel<<<g, 256, 0, stream>>>(Wq, WqT, DIM, DIM);
    tr_cvt_kernel<<<g, 256, 0, stream>>>(Wk, WkT, DIM, DIM);
    tr_cvt_kernel<<<g, 256, 0, stream>>>(Wv, WvT, DIM, DIM);
    tr_cvt_kernel<<<g, 256, 0, stream>>>(Wsplit, WspT, DIM, DIM);
    tr_cvt_kernel<<<g, 256, 0, stream>>>(Wmerge, WmgT, DIM, DIM);
    int m = DH * HID, gm = (m + 255) / 256;
    tr_cvt_kernel<<<gm, 256, 0, stream>>>(mem_w0, w0t, DH, HID);   // w0t[h][d]
    tr_cvt_kernel<<<gm, 256, 0, stream>>>(mem_w1, w1t, HID, DH);   // w1t[d][h]
    cvt_kernel   <<<gm, 256, 0, stream>>>(mem_w1, w1n, m);         // w1n[h][d]
  }

  // ---- layernorms + adaptive lr ----
  ln_kernel<<<ROWS, 256, 0, stream>>>(seq, g_store, b_store, g_ret, b_ret,
                                      W_step, b_step, s_bf, r_bf, lr);

  // ---- dense 768 projections (M=16384, N=768, K=768) ----
  const dim3 gP(ROWS / 32, DIM / 32 / 4, 1);           // (512, 6, 1)
  // k chain
  gemm_wmma<true, true, EPI_BF><<<gP, blk, 0, stream>>>(
      s_bf, 0, 0, DIM, 1, WkT, 0, 0, 1, DIM, tmp_bf, 0, 0, DIM, DIM, DIM,
      nullptr, 0, 0, 0, nullptr, 0, 0);
  gemm_wmma<true, true, EPI_BF><<<gP, blk, 0, stream>>>(
      tmp_bf, 0, 0, DIM, 1, WspT, 0, 0, 1, DIM, kproj, 0, 0, DIM, DIM, DIM,
      nullptr, 0, 0, 0, nullptr, 0, 0);
  // v chain
  gemm_wmma<true, true, EPI_BF><<<gP, blk, 0, stream>>>(
      s_bf, 0, 0, DIM, 1, WvT, 0, 0, 1, DIM, tmp_bf, 0, 0, DIM, DIM, DIM,
      nullptr, 0, 0, 0, nullptr, 0, 0);
  gemm_wmma<true, true, EPI_BF><<<gP, blk, 0, stream>>>(
      tmp_bf, 0, 0, DIM, 1, WspT, 0, 0, 1, DIM, vproj, 0, 0, DIM, DIM, DIM,
      nullptr, 0, 0, 0, nullptr, 0, 0);
  // q chain
  gemm_wmma<true, true, EPI_BF><<<gP, blk, 0, stream>>>(
      r_bf, 0, 0, DIM, 1, WqT, 0, 0, 1, DIM, tmp_bf, 0, 0, DIM, DIM, DIM,
      nullptr, 0, 0, 0, nullptr, 0, 0);
  gemm_wmma<true, true, EPI_BF><<<gP, blk, 0, stream>>>(
      tmp_bf, 0, 0, DIM, 1, WspT, 0, 0, 1, DIM, qproj, 0, 0, DIM, DIM, DIM,
      nullptr, 0, 0, 0, nullptr, 0, 0);

  // ---- pred0 = clip(kf @ w0)  [bh: M=2048, N=384, K=96] ----
  gemm_wmma<true, true, EPI_CLIP><<<dim3(NSEQ / 32, HID / 32 / 4, 64), blk, 0, stream>>>(
      kproj, (size_t)NSEQ * DIM, DH, DIM, 1,
      w0t, 0, 0, 1, DH,
      pred0, (size_t)8 * NSEQ * HID, (size_t)NSEQ * HID, HID, HID, DH,
      nullptr, 0, 0, 0, nullptr, 0, 0);

  // ---- we = (clip(pred0 @ w1) - vf) * lr  [M=2048, N=96, K=384] ----
  gemm_wmma<true, true, EPI_WE><<<dim3(NSEQ / 32, 1, 64), blk, 0, stream>>>(
      pred0, (size_t)8 * NSEQ * HID, (size_t)NSEQ * HID, HID, 1,
      w1t, 0, 0, 1, HID,
      we, (size_t)8 * NSEQ * DH, (size_t)NSEQ * DH, DH, DH, HID,
      vproj, (size_t)NSEQ * DIM, DH, DIM,
      lr, (size_t)NSEQ * NH, 1);

  // ---- gw1 = pred0^T @ we ; w1_new^T stored  [M=384, N=96, K=2048] ----
  gemm_wmma<false, false, EPI_WSUBT><<<dim3(HID / 32, 1, 64), blk, 0, stream>>>(
      pred0, (size_t)8 * NSEQ * HID, (size_t)NSEQ * HID, 1, HID,
      we, (size_t)8 * NSEQ * DH, (size_t)NSEQ * DH, DH, 1,
      w1newT, 8 * WG, WG, HID, DH, NSEQ,
      w1t, 0, 0, 0, nullptr, 0, 0);

  // ---- dh0 = we @ w1^T  [M=2048, N=384, K=96] ----
  gemm_wmma<true, true, EPI_BF><<<dim3(NSEQ / 32, HID / 32 / 4, 64), blk, 0, stream>>>(
      we, (size_t)8 * NSEQ * DH, (size_t)NSEQ * DH, DH, 1,
      w1n, 0, 0, 1, DH,
      dh0, (size_t)8 * NSEQ * HID, (size_t)NSEQ * HID, HID, HID, DH,
      nullptr, 0, 0, 0, nullptr, 0, 0);

  // ---- gw0 = kf^T @ dh0 ; w0_new^T stored  [M=96, N=384, K=2048] ----
  gemm_wmma<false, false, EPI_WSUBT><<<dim3(DH / 32, HID / 32 / 4, 64), blk, 0, stream>>>(
      kproj, (size_t)NSEQ * DIM, DH, 1, DIM,
      dh0, (size_t)8 * NSEQ * HID, (size_t)NSEQ * HID, HID, 1,
      w0newT, 8 * WG, WG, DH, HID, NSEQ,
      w0t, 0, 0, 0, nullptr, 0, 0);

  // ---- h = gelu(q @ w0_new)  [M=2048, N=384, K=96] ----
  gemm_wmma<true, true, EPI_GELU><<<dim3(NSEQ / 32, HID / 32 / 4, 64), blk, 0, stream>>>(
      qproj, (size_t)NSEQ * DIM, DH, DIM, 1,
      w0newT, 8 * WG, WG, 1, DH,
      h_bf, (size_t)8 * NSEQ * HID, (size_t)NSEQ * HID, HID, HID, DH,
      nullptr, 0, 0, 0, nullptr, 0, 0);

  // ---- merged = h @ w1_new (head-merge scatter)  [M=2048, N=96, K=384] ----
  gemm_wmma<true, true, EPI_BF><<<dim3(NSEQ / 32, 1, 64), blk, 0, stream>>>(
      h_bf, (size_t)8 * NSEQ * HID, (size_t)NSEQ * HID, HID, 1,
      w1newT, 8 * WG, WG, 1, HID,
      merged, (size_t)NSEQ * DIM, DH, DIM, DH, HID,
      nullptr, 0, 0, 0, nullptr, 0, 0);

  // ---- out = merged @ Wmerge  [M=16384, N=768, K=768], f32 output ----
  gemm_wmma<true, true, EPI_F32><<<gP, blk, 0, stream>>>(
      merged, 0, 0, DIM, 1, WmgT, 0, 0, 1, DIM, out, 0, 0, DIM, DIM, DIM,
      nullptr, 0, 0, 0, nullptr, 0, 0);

  (void)in_sizes; (void)n_in; (void)out_size; (void)ws_size;
}
